// SelfAttention_13838384628048
// MI455X (gfx1250) — matrix operand order repaired and checked by reading, not compiled, not run on hardware
//
#include <hip/hip_runtime.h>
#include <hip/hip_bf16.h>

typedef __attribute__((ext_vector_type(16))) _Float16 v16h;
typedef __attribute__((ext_vector_type(8)))  _Float16 v8h;
typedef __attribute__((ext_vector_type(8)))  float    v8f;
typedef __attribute__((ext_vector_type(4)))  float    v4f;

#define D_MODEL   1024
#define N_HEADS   16
#define HEAD_DIM  64
#define SEQ       2048
#define BATCH     4
// 1/sqrt(64) * log2(e): softmax computed in exp2 domain (v_exp_f32 is exp2)
#define SM_SCALE_LOG2E 0.180336880260832f

// ---------------- WMMA wrapper ----------------
static __device__ __forceinline__ v8f wmma_f16(v16h a, v16h b, v8f c) {
  return __builtin_amdgcn_wmma_f32_16x16x32_f16(false, a, false, b,
                                                (short)0, c, false, false);
}

// ---------------- VALU cross-lane xor reduction (DPP16 ROW_XMASK) ----------
// Keeps reductions on the VALU pipe (co-executes with XDL WMMA) instead of
// ds_bpermute. ROW_XMASK = 0x160 | mask: lane <-> lane^mask within 16 lanes.
template <int MASK>
static __device__ __forceinline__ float dpp_xor(float x) {
  int xi = __float_as_int(x);
  int yi = __builtin_amdgcn_update_dpp(xi, xi, 0x160 | MASK, 0xf, 0xf, true);
  return __int_as_float(yi);
}
static __device__ __forceinline__ float rowmax16(float x) {
  x = fmaxf(x, dpp_xor<1>(x));
  x = fmaxf(x, dpp_xor<2>(x));
  x = fmaxf(x, dpp_xor<4>(x));
  x = fmaxf(x, dpp_xor<8>(x));
  return x;
}
static __device__ __forceinline__ float rowsum16(float x) {
  x += dpp_xor<1>(x);
  x += dpp_xor<2>(x);
  x += dpp_xor<4>(x);
  x += dpp_xor<8>(x);
  return x;
}

// ---------------- fragment loaders ----------------
// A-matrix 16x32 f16: lane L -> row m=L&15, hi=L>>4;
// VGPR0-3: K = hi*8 + 0..7 ; VGPR4-7: K = 16 + hi*8 + 0..7   (ISA 7.12.2)
static __device__ __forceinline__ v16h afrag_f32(const float* __restrict__ tile,
                                                 int ld, int lane) {
  const int row = lane & 15, hi = lane >> 4;
  const float* p = tile + (size_t)row * ld + hi * 8;
  v4f a0 = *(const v4f*)(p);
  v4f a1 = *(const v4f*)(p + 4);
  v4f a2 = *(const v4f*)(p + 16);
  v4f a3 = *(const v4f*)(p + 20);
  v16h r;
#pragma unroll
  for (int i = 0; i < 4; ++i) {
    r[i]      = (_Float16)a0[i];
    r[4 + i]  = (_Float16)a1[i];
    r[8 + i]  = (_Float16)a2[i];
    r[12 + i] = (_Float16)a3[i];
  }
  return r;
}

static __device__ __forceinline__ v16h afrag_f16(const _Float16* __restrict__ tile,
                                                 int ld, int lane) {
  const int row = lane & 15, hi = lane >> 4;
  const _Float16* p = tile + (size_t)row * ld + hi * 8;
  v8h lo = *(const v8h*)(p);
  v8h hh = *(const v8h*)(p + 16);
  v16h r;
#pragma unroll
  for (int i = 0; i < 8; ++i) { r[i] = lo[i]; r[8 + i] = hh[i]; }
  return r;
}

// B-matrix 32x16 f16: lane L -> col n=L&15, K = (L>>4)*8 + 0..7, 16 + (L>>4)*8 + 0..7.
// B[k][n] = src[n*ld + k] (src holds B transposed, rows contiguous over k).
static __device__ __forceinline__ v16h bfrag_f32(const float* __restrict__ src,
                                                 int ld, int lane) {
  const int n = lane & 15, koff = (lane >> 4) * 8;
  const float* p = src + (size_t)n * ld + koff;
  v4f a0 = *(const v4f*)(p);
  v4f a1 = *(const v4f*)(p + 4);
  v4f a2 = *(const v4f*)(p + 16);
  v4f a3 = *(const v4f*)(p + 20);
  v16h r;
#pragma unroll
  for (int i = 0; i < 4; ++i) {
    r[i]      = (_Float16)a0[i];
    r[4 + i]  = (_Float16)a1[i];
    r[8 + i]  = (_Float16)a2[i];
    r[12 + i] = (_Float16)a3[i];
  }
  return r;
}

static __device__ __forceinline__ v16h bfrag_f16(const _Float16* __restrict__ src,
                                                 int ld, int lane) {
  const int n = lane & 15, koff = (lane >> 4) * 8;
  const _Float16* p = src + (size_t)n * ld + koff;
  v8h lo = *(const v8h*)(p);
  v8h hh = *(const v8h*)(p + 16);
  v16h r;
#pragma unroll
  for (int i = 0; i < 8; ++i) { r[i] = lo[i]; r[8 + i] = hh[i]; }
  return r;
}

// ---------------- kernel 1: QKV projection (f32 x, W -> f16) ----------------
// Y = x @ W^T, wave computes a 32x64 strip (B-frags reused across 2 row tiles).
// MODE 0: store [B,H,S,Dh]; MODE 1: store transposed [B,H,Dh,S] (for V).
// Grid: 4096 wave-jobs = 512 blocks x 8 waves.
template <int MODE>
__global__ __launch_bounds__(256) void proj_kernel(const float* __restrict__ X,
                                                   const float* __restrict__ W,
                                                   _Float16* __restrict__ Y) {
  const int wave = threadIdx.x >> 5, lane = threadIdx.x & 31;
  const int job = blockIdx.x * 8 + wave;          // [0, 4096)
  const int m0 = (job >> 4) * 32;                 // row tile over B*S
  const int tn = (job & 15) * 64;                 // col base over D_MODEL
  const int hi = lane >> 4, nn = lane & 15;

  v8f acc[2][4] = {};
  for (int k = 0; k < D_MODEL; k += 32) {
    v16h a0 = afrag_f32(X + (size_t)m0 * D_MODEL + k, D_MODEL, lane);
    v16h a1 = afrag_f32(X + (size_t)(m0 + 16) * D_MODEL + k, D_MODEL, lane);
#pragma unroll
    for (int j = 0; j < 4; ++j) {
      v16h b = bfrag_f32(W + (size_t)(tn + j * 16) * D_MODEL + k, D_MODEL, lane);
      acc[0][j] = wmma_f16(a0, b, acc[0][j]);
      acc[1][j] = wmma_f16(a1, b, acc[1][j]);
    }
  }

  const int bb = m0 >> 11;                        // batch (rows of 2048)
#pragma unroll
  for (int t = 0; t < 2; ++t) {
    const int mt = m0 + t * 16;
    if (MODE == 0) {
#pragma unroll
      for (int j = 0; j < 4; ++j) {
        const int n = tn + j * 16 + nn;
        const int h = n >> 6, dh = n & 63;
#pragma unroll
        for (int r = 0; r < 8; ++r) {
          const int s = (mt & 2047) + r + 8 * hi;
          Y[(((size_t)bb * N_HEADS + h) * SEQ + s) * HEAD_DIM + dh] =
              (_Float16)acc[t][j][r];
        }
      }
    } else {
      const int s0 = (mt & 2047) + 8 * hi;        // 8 contiguous rows -> b128
#pragma unroll
      for (int j = 0; j < 4; ++j) {
        const int n = tn + j * 16 + nn;
        const int h = n >> 6, dh = n & 63;
        v8h pk;
#pragma unroll
        for (int r = 0; r < 8; ++r) pk[r] = (_Float16)acc[t][j][r];
        *(v8h*)(Y + (((size_t)bb * N_HEADS + h) * HEAD_DIM + dh) * SEQ + s0) = pk;
      }
    }
  }
}

// ---------------- kernel 2: flash attention ----------------
// One wave per (b, h, 32-query tile); keys streamed in chunks of 32 with
// online softmax (exp2 domain). Each K/V fragment feeds two WMMAs.
__global__ __launch_bounds__(256) void flash_kernel(const _Float16* __restrict__ Q,
                                                    const _Float16* __restrict__ K,
                                                    const _Float16* __restrict__ Vt,
                                                    _Float16* __restrict__ AO) {
  __shared__ _Float16 pshare[8][32 * 32];         // 2KB per wave
  const int wave = threadIdx.x >> 5, lane = threadIdx.x & 31;
  const int job = blockIdx.x * 8 + wave;          // [0, 4096)
  const int qt = job & 63;                        // 32-query tile
  const int h  = (job >> 6) & 15;
  const int bb = job >> 10;
  const size_t bh = (size_t)bb * N_HEADS + h;
  const int hi = lane >> 4, nn = lane & 15;

  const _Float16* qbase = Q + (bh * SEQ + (size_t)qt * 32) * HEAD_DIM;
  v16h aq[2][2];
#pragma unroll
  for (int t = 0; t < 2; ++t) {
    aq[t][0] = afrag_f16(qbase + (size_t)t * 16 * HEAD_DIM, HEAD_DIM, lane);
    aq[t][1] = afrag_f16(qbase + (size_t)t * 16 * HEAD_DIM + 32, HEAD_DIM, lane);
  }

  v8f o[2][4] = {};
  float mrow[2][8], lrow[2][8];
#pragma unroll
  for (int t = 0; t < 2; ++t)
#pragma unroll
    for (int r = 0; r < 8; ++r) { mrow[t][r] = -1e30f; lrow[t][r] = 0.0f; }
  _Float16* lds = pshare[wave];

  const _Float16* kb0 = K + bh * SEQ * HEAD_DIM;
  const _Float16* vb0 = Vt + bh * (size_t)HEAD_DIM * SEQ;

  for (int kc = 0; kc < SEQ; kc += 32) {
    const _Float16* kbase = kb0 + (size_t)kc * HEAD_DIM;
    __builtin_prefetch(kbase + 32 * HEAD_DIM, 0, 3);  // next K tile
    // K fragments loaded once, reused by both query row-tiles.
    const v16h bk00 = bfrag_f16(kbase, HEAD_DIM, lane);                 // keys 0-15,  dh 0-31
    const v16h bk01 = bfrag_f16(kbase + 32, HEAD_DIM, lane);            // keys 0-15,  dh 32-63
    const v16h bk10 = bfrag_f16(kbase + 16 * HEAD_DIM, HEAD_DIM, lane); // keys 16-31, dh 0-31
    const v16h bk11 = bfrag_f16(kbase + 16 * HEAD_DIM + 32, HEAD_DIM, lane);

    float alpha[2][8];
#pragma unroll
    for (int t = 0; t < 2; ++t) {
      v8f s0 = {}, s1 = {};
      s0 = wmma_f16(aq[t][0], bk00, s0);
      s0 = wmma_f16(aq[t][1], bk01, s0);
      s1 = wmma_f16(aq[t][0], bk10, s1);
      s1 = wmma_f16(aq[t][1], bk11, s1);
#pragma unroll
      for (int r = 0; r < 8; ++r) {
        const float x0 = s0[r] * SM_SCALE_LOG2E;
        const float x1 = s1[r] * SM_SCALE_LOG2E;
        const float vmax = rowmax16(fmaxf(x0, x1));
        const float mnew = fmaxf(mrow[t][r], vmax);
        alpha[t][r] = __builtin_amdgcn_exp2f(mrow[t][r] - mnew);
        mrow[t][r] = mnew;
        const float p0 = __builtin_amdgcn_exp2f(x0 - mnew);
        const float p1 = __builtin_amdgcn_exp2f(x1 - mnew);
        lrow[t][r] = lrow[t][r] * alpha[t][r] + rowsum16(p0 + p1);
        lds[(t * 16 + r + 8 * hi) * 32 + nn]      = (_Float16)p0;
        lds[(t * 16 + r + 8 * hi) * 32 + 16 + nn] = (_Float16)p1;
      }
#pragma unroll
      for (int j = 0; j < 4; ++j)
#pragma unroll
        for (int r = 0; r < 8; ++r) o[t][j][r] *= alpha[t][r];
    }

    asm volatile("s_wait_dscnt 0" ::: "memory");   // P visible before re-read
    const v16h pa0 = afrag_f16(lds, 32, lane);           // rows 0-15
    const v16h pa1 = afrag_f16(lds + 16 * 32, 32, lane); // rows 16-31
    const _Float16* vb = vb0 + kc;                       // Vt[dh][s]
#pragma unroll
    for (int j = 0; j < 4; ++j) {
      const v16h bv = bfrag_f16(vb + (size_t)(j * 16) * SEQ, SEQ, lane);
      o[0][j] = wmma_f16(pa0, bv, o[0][j]);
      o[1][j] = wmma_f16(pa1, bv, o[1][j]);
    }
  }

  // epilogue: normalize, pack heads back into [B,S,D_MODEL] f16
#pragma unroll
  for (int t = 0; t < 2; ++t)
#pragma unroll
    for (int j = 0; j < 4; ++j) {
      const int n = h * 64 + j * 16 + nn;
#pragma unroll
      for (int r = 0; r < 8; ++r) {
        const int s = qt * 32 + t * 16 + r + 8 * hi;
        AO[((size_t)bb * SEQ + s) * D_MODEL + n] =
            (_Float16)(o[t][j][r] / lrow[t][r]);
      }
    }
}

// ---------------- kernel 3: output projection + bias ----------------
__global__ __launch_bounds__(256) void oproj_kernel(const _Float16* __restrict__ AO,
                                                    const float* __restrict__ Wo,
                                                    const float* __restrict__ bo,
                                                    float* __restrict__ out) {
  const int wave = threadIdx.x >> 5, lane = threadIdx.x & 31;
  const int job = blockIdx.x * 8 + wave;          // [0, 4096)
  const int m0 = (job >> 4) * 32;
  const int tn = (job & 15) * 64;
  const int hi = lane >> 4, nn = lane & 15;

  v8f acc[2][4] = {};
  for (int k = 0; k < D_MODEL; k += 32) {
    v16h a0 = afrag_f16(AO + (size_t)m0 * D_MODEL + k, D_MODEL, lane);
    v16h a1 = afrag_f16(AO + (size_t)(m0 + 16) * D_MODEL + k, D_MODEL, lane);
#pragma unroll
    for (int j = 0; j < 4; ++j) {
      v16h b = bfrag_f32(Wo + (size_t)(tn + j * 16) * D_MODEL + k, D_MODEL, lane);
      acc[0][j] = wmma_f16(a0, b, acc[0][j]);
      acc[1][j] = wmma_f16(a1, b, acc[1][j]);
    }
  }
#pragma unroll
  for (int t = 0; t < 2; ++t)
#pragma unroll
    for (int j = 0; j < 4; ++j) {
      const int n = tn + j * 16 + nn;
      const float bias = bo[n];
#pragma unroll
      for (int r = 0; r < 8; ++r)
        out[(size_t)(m0 + t * 16 + r + 8 * hi) * D_MODEL + n] = acc[t][j][r] + bias;
    }
}

// ---------------- host launch ----------------
extern "C" void kernel_launch(void* const* d_in, const int* in_sizes, int n_in,
                              void* d_out, int out_size, void* d_ws, size_t ws_size,
                              hipStream_t stream) {
  (void)in_sizes; (void)n_in; (void)out_size; (void)ws_size;
  const float* x  = (const float*)d_in[0];
  const float* Wq = (const float*)d_in[1];
  const float* Wk = (const float*)d_in[2];
  const float* Wv = (const float*)d_in[3];
  const float* Wo = (const float*)d_in[4];
  const float* bo = (const float*)d_in[5];
  float* out = (float*)d_out;

  const size_t ELEMS = (size_t)BATCH * N_HEADS * SEQ * HEAD_DIM;  // 8,388,608
  _Float16* q  = (_Float16*)d_ws;
  _Float16* k  = q + ELEMS;
  _Float16* vt = k + ELEMS;
  _Float16* ao = vt + ELEMS;

  dim3 block(256), grid(512);    // 4096 wave-jobs each phase
  proj_kernel<0><<<grid, block, 0, stream>>>(x, Wq, q);
  proj_kernel<0><<<grid, block, 0, stream>>>(x, Wk, k);
  proj_kernel<1><<<grid, block, 0, stream>>>(x, Wv, vt);
  flash_kernel<<<grid, block, 0, stream>>>(q, k, vt, ao);
  oproj_kernel<<<grid, block, 0, stream>>>(ao, Wo, bo, out);
}
